// BiMambaMSM_9620726743767
// MI455X (gfx1250) — compile-verified, matrix operand order checked by reading, not run on hardware
//
#include <hip/hip_runtime.h>
#include <math.h>

// ---------------------------------------------------------------------------
// BiMamba2 masked-sequence model forward for MI455X (gfx1250, wave32, WMMA)
// ---------------------------------------------------------------------------
// Shapes: B=4, T=1024, D_IN=80, D_MODEL=768, D_INNER=1536, D_STATE=16,
// NHEADS=24, HEADDIM=64, CONV_DIM=1568, D_IN_PROJ=3128 (padded to 3200).
// Heavy GEMMs: v_wmma_f32_16x16x32_bf16 with fp32 accumulation, fed by a
// double-buffered global_load_async_to_lds_b128 pipeline (ASYNCcnt).
// ---------------------------------------------------------------------------

typedef __bf16 bf16_t;
typedef __attribute__((ext_vector_type(8)))  bf16_t v8bf;
typedef __attribute__((ext_vector_type(16))) bf16_t v16bf;
typedef __attribute__((ext_vector_type(8)))  float  v8f;

#define NROWS   4096          // B*T
#define DMODEL  768
#define DINNER  1536
#define DSTATE  16
#define NHEADS  24
#define HEADDIM 64
#define CONVDIM 1568
#define NPROJ   3128
#define NPROJP  3200          // padded to multiple of 128
#define TLEN    1024

static __device__ __forceinline__ bf16_t to_bf16(float f) {
    unsigned u = __builtin_bit_cast(unsigned, f);
    unsigned r = u + 0x7FFFu + ((u >> 16) & 1u);   // round-to-nearest-even
    unsigned short h = (unsigned short)(r >> 16);
    return __builtin_bit_cast(bf16_t, h);
}

static __device__ __forceinline__ float sigmoidf_(float x) {
    return 1.0f / (1.0f + __expf(-x));
}

// LDS byte offset of a __shared__ object: generic LDS addresses carry the
// wave-relative byte offset in addr[31:0] (CDNA5 aperture mapping).
static __device__ __forceinline__ unsigned lds_off(const void* p) {
    return (unsigned)(unsigned long long)(uintptr_t)p;
}

// Async Global->LDS DMA: 2 x 16B per call (INST_OFFSET applies to both the
// LDS and the global address, so one address pair covers both chunks).
static __device__ __forceinline__ void async_copy_32B(unsigned lds, const void* gaddr) {
    asm volatile(
        "global_load_async_to_lds_b128 %0, %1, off\n\t"
        "global_load_async_to_lds_b128 %0, %1, off offset:16"
        :: "v"(lds), "v"(gaddr) : "memory");
}

// ---------------------------------------------------------------------------
// Weight conversion: fp32 (K x N, row-major) -> bf16 W^T (Npad x K), zero pad.
// Pre-transposing makes the GEMM B-fragment loads contiguous along K.
// ---------------------------------------------------------------------------
__global__ void convert_wt_kernel(const float* __restrict__ src, bf16_t* __restrict__ dst,
                                  int K, int N, int Npad) {
    int idx = blockIdx.x * blockDim.x + threadIdx.x;
    if (idx >= Npad * K) return;
    int k = idx % K;
    int n = idx / K;
    float v = (n < N) ? src[(size_t)k * N + n] : 0.0f;
    dst[idx] = to_bf16(v);
}

// ---------------------------------------------------------------------------
// Tiled bf16 WMMA GEMM:  C[M x Npad] (+)= A[M x K] * Bt[Npad x K]^T (+ bias)
// Block tile 128x128, K-step 32, 256 threads = 8 waves arranged 2(M) x 4(N),
// each wave computes 4x2 grid of 16x16 tiles => 8 v_wmma per K-step.
// Staging: double-buffered global_load_async_to_lds_b128 (4 DMAs per thread
// per tile), drained with s_wait_asynccnt so the next tile streams in while
// the current one is consumed by the WMMAs.
// ---------------------------------------------------------------------------
__global__ __launch_bounds__(256)
void gemm_bf16_kernel(const bf16_t* __restrict__ A,   // M x K row-major
                      const bf16_t* __restrict__ Bt,  // Npad x K row-major (B^T)
                      const float* __restrict__ bias, // length >= Nreal or null
                      float* __restrict__ C,          // M x ldc row-major
                      int M, int Npad, int Nreal, int K, int ldc, int accumulate)
{
    __shared__ bf16_t sA[2][128][40];   // 32 K-cols + 8 pad (80B row stride)
    __shared__ bf16_t sB[2][128][40];

    const int tid   = threadIdx.x;
    const int wave  = tid >> 5;
    const int lane  = tid & 31;
    const int waveM = wave >> 2;              // 0..1  (64 rows each)
    const int waveN = wave & 3;               // 0..3  (32 cols each)
    const int blockM = blockIdx.x * 128;
    const int blockN = blockIdx.y * 128;

    const int lr = lane & 15;                 // row/col within a 16-tile
    const int lh = lane >> 4;                 // half select (K halves / M halves)

    v8f acc[4][2] = {};

    const int ldRow = tid >> 1;               // 0..127
    const int ldCol = (tid & 1) << 4;         // 0 or 16

    const bf16_t* gA = A  + (size_t)(blockM + ldRow) * K + ldCol;
    const bf16_t* gB = Bt + (size_t)(blockN + ldRow) * K + ldCol;

    const unsigned la0 = lds_off(&sA[0][ldRow][ldCol]);
    const unsigned lb0 = lds_off(&sB[0][ldRow][ldCol]);
    const unsigned la1 = lds_off(&sA[1][ldRow][ldCol]);
    const unsigned lb1 = lds_off(&sB[1][ldRow][ldCol]);

    const int KT = K >> 5;                    // K / 32 tiles

    // Prologue: stage tile 0 into buffer 0.
    async_copy_32B(la0, gA);
    async_copy_32B(lb0, gB);

    for (int kt = 0; kt < KT; ++kt) {
        const int cur = kt & 1;

        if (kt + 1 < KT) {
            // Stream next tile into the other buffer, then wait only for the
            // current tile's 4 DMAs (the 4 just issued stay in flight).
            const bf16_t* nA = gA + (kt + 1) * 32;
            const bf16_t* nB = gB + (kt + 1) * 32;
            async_copy_32B(cur ? la0 : la1, nA);
            async_copy_32B(cur ? lb0 : lb1, nB);
            asm volatile("s_wait_asynccnt 0x4" ::: "memory");
        } else {
            asm volatile("s_wait_asynccnt 0x0" ::: "memory");
        }
        __syncthreads();   // current tile visible to all waves

        // A fragment (16x32 bf16): lane lr = M row; halves of K split at lane 16.
        v16bf aF[4], bF[2];
        #pragma unroll
        for (int m = 0; m < 4; ++m) {
            const bf16_t* p = &sA[cur][waveM * 64 + m * 16 + lr][lh * 8];
            v8bf lo = *(const v8bf*)p;
            v8bf hi = *(const v8bf*)(p + 16);
            aF[m] = __builtin_shufflevector(lo, hi, 0,1,2,3,4,5,6,7,8,9,10,11,12,13,14,15);
        }
        // B fragment (32x16 bf16): Bt stored N-major so same loader as A.
        #pragma unroll
        for (int n = 0; n < 2; ++n) {
            const bf16_t* p = &sB[cur][waveN * 32 + n * 16 + lr][lh * 8];
            v8bf lo = *(const v8bf*)p;
            v8bf hi = *(const v8bf*)(p + 16);
            bF[n] = __builtin_shufflevector(lo, hi, 0,1,2,3,4,5,6,7,8,9,10,11,12,13,14,15);
        }

        #pragma unroll
        for (int m = 0; m < 4; ++m)
            #pragma unroll
            for (int n = 0; n < 2; ++n)
                acc[m][n] = __builtin_amdgcn_wmma_f32_16x16x32_bf16(
                    false, aF[m], false, bF[n], (short)0, acc[m][n], false, false);

        // All waves done reading buffer `cur` (ds reads were consumed by the
        // WMMAs above) before the next iteration's DMA overwrites it.
        __syncthreads();
    }

    // Epilogue. D layout: N = lane&15; M = (lane>>4)*8 + vgpr.
    const int mhalf = lh * 8;
    #pragma unroll
    for (int m = 0; m < 4; ++m) {
        #pragma unroll
        for (int n = 0; n < 2; ++n) {
            const int gn = blockN + waveN * 32 + n * 16 + lr;
            const float bi = (bias != nullptr && gn < Nreal) ? bias[gn] : 0.0f;
            #pragma unroll
            for (int r = 0; r < 8; ++r) {
                const int gm = blockM + waveM * 64 + m * 16 + mhalf + r;
                const size_t off = (size_t)gm * ldc + gn;
                float v = acc[m][n][r] + bi;
                if (accumulate) v += C[off];
                C[off] = v;
            }
        }
    }
}

// ---------------------------------------------------------------------------
// Input projection (80 -> 768) + mask-token substitution. fp32 (tiny GEMM).
// ---------------------------------------------------------------------------
__global__ void proj_in_kernel(const float* __restrict__ x, const int* __restrict__ mask,
                               const float* __restrict__ W, const float* __restrict__ b,
                               const float* __restrict__ mtok, float* __restrict__ H)
{
    int idx = blockIdx.x * blockDim.x + threadIdx.x;
    if (idx >= NROWS * DMODEL) return;
    int c = idx % DMODEL;
    int row = idx / DMODEL;
    float v;
    if (mask[row] != 0) {
        v = mtok[c];
    } else {
        v = b[c];
        const float* xr = x + (size_t)row * 80;
        #pragma unroll 8
        for (int k = 0; k < 80; ++k) v += xr[k] * W[(size_t)k * DMODEL + c];
    }
    H[idx] = v;
}

// ---------------------------------------------------------------------------
// LayerNorm over 768 cols. Optionally emits bf16 (GEMM input) and/or fp32.
// One block per row, 256 threads x 3 elements.
// ---------------------------------------------------------------------------
__global__ __launch_bounds__(256)
void layernorm768_kernel(const float* __restrict__ X, const float* __restrict__ w,
                         const float* __restrict__ b, bf16_t* __restrict__ out_bf,
                         float* __restrict__ out_f32)
{
    __shared__ float red[256];
    const int row = blockIdx.x;
    const int tid = threadIdx.x;
    const float* xr = X + (size_t)row * DMODEL;

    float vals[3];
    float s = 0.0f;
    #pragma unroll
    for (int i = 0; i < 3; ++i) { vals[i] = xr[tid + i * 256]; s += vals[i]; }
    red[tid] = s;
    __syncthreads();
    for (int off = 128; off > 0; off >>= 1) {
        if (tid < off) red[tid] += red[tid + off];
        __syncthreads();
    }
    const float mu = red[0] * (1.0f / DMODEL);
    __syncthreads();

    float s2 = 0.0f;
    #pragma unroll
    for (int i = 0; i < 3; ++i) { float d = vals[i] - mu; s2 += d * d; }
    red[tid] = s2;
    __syncthreads();
    for (int off = 128; off > 0; off >>= 1) {
        if (tid < off) red[tid] += red[tid + off];
        __syncthreads();
    }
    const float inv = rsqrtf(red[0] * (1.0f / DMODEL) + 1e-5f);

    #pragma unroll
    for (int i = 0; i < 3; ++i) {
        const int c = tid + i * 256;
        const float o = (vals[i] - mu) * inv * w[c] + b[c];
        if (out_bf)  out_bf[(size_t)row * DMODEL + c] = to_bf16(o);
        if (out_f32) out_f32[(size_t)row * DMODEL + c] = o;
    }
}

// ---------------------------------------------------------------------------
// dt = softplus(raw + dt_bias), dA = exp(-exp(A_log) * dt). Interleaved store.
// ---------------------------------------------------------------------------
__global__ void dtda_kernel(const float* __restrict__ ZX, const float* __restrict__ dt_bias,
                            const float* __restrict__ A_log, float* __restrict__ DTDA)
{
    int idx = blockIdx.x * blockDim.x + threadIdx.x;
    if (idx >= NROWS * NHEADS) return;
    int h = idx % NHEADS;
    int row = idx / NHEADS;
    float raw = ZX[(size_t)row * NPROJP + DINNER + CONVDIM + h] + dt_bias[h];
    float dt = (raw > 20.0f) ? raw : log1pf(__expf(raw));
    float dA = __expf(-__expf(A_log[h]) * dt);
    DTDA[(size_t)row * (2 * NHEADS) + h * 2 + 0] = dt;
    DTDA[(size_t)row * (2 * NHEADS) + h * 2 + 1] = dA;
}

// ---------------------------------------------------------------------------
// Depthwise causal conv (K=7) + SiLU. backward=1 flips the time taps, which
// implements conv-on-reversed-sequence without materializing the reversal.
// ---------------------------------------------------------------------------
__global__ void conv_silu_kernel(const float* __restrict__ ZX, const float* __restrict__ w,
                                 const float* __restrict__ bias, float* __restrict__ XC,
                                 int backward)
{
    int idx = blockIdx.x * blockDim.x + threadIdx.x;
    if (idx >= NROWS * CONVDIM) return;
    int c = idx % CONVDIM;
    int row = idx / CONVDIM;
    int t = row % TLEN;
    int b = row / TLEN;
    float s = bias[c];
    #pragma unroll
    for (int j = 0; j < 7; ++j) {
        int tj = backward ? (t + 6 - j) : (t + j - 6);
        if (tj >= 0 && tj < TLEN)
            s += w[c * 7 + j] * ZX[((size_t)(b * TLEN + tj)) * NPROJP + DINNER + c];
    }
    XC[(size_t)row * CONVDIM + c] = s * sigmoidf_(s);   // SiLU
}

// ---------------------------------------------------------------------------
// Sequential SSM scan. One thread per (head, headdim) lane keeps the 16-float
// state in VGPRs. grid = (batch=4, headgroup=3), block = 512 (8 heads x 64 p).
// ---------------------------------------------------------------------------
__global__ __launch_bounds__(512)
void scan_kernel(const float* __restrict__ XC, const float* __restrict__ DTDA,
                 const float* __restrict__ Dp, float* __restrict__ Y, int backward)
{
    const int b    = blockIdx.x;                 // 0..3
    const int hg   = blockIdx.y;                 // 0..2
    const int hl   = threadIdx.x >> 6;           // 0..7
    const int p    = threadIdx.x & 63;           // 0..63
    const int head = hg * 8 + hl;

    float h[DSTATE];
    #pragma unroll
    for (int n = 0; n < DSTATE; ++n) h[n] = 0.0f;
    const float D = Dp[head];

    for (int tt = 0; tt < TLEN; ++tt) {
        const int t = backward ? (TLEN - 1 - tt) : tt;
        const size_t rb = (size_t)(b * TLEN + t);
        const float dt = DTDA[rb * (2 * NHEADS) + head * 2 + 0];
        const float dA = DTDA[rb * (2 * NHEADS) + head * 2 + 1];
        const float* xr = XC + rb * CONVDIM;
        const float x  = xr[head * HEADDIM + p];
        const float* Bv = xr + DINNER;
        const float* Cv = xr + DINNER + DSTATE;
        const float dtx = dt * x;
        float y = 0.0f;
        #pragma unroll
        for (int n = 0; n < DSTATE; ++n) {
            h[n] = h[n] * dA + dtx * Bv[n];
            y += h[n] * Cv[n];
        }
        Y[rb * DINNER + head * HEADDIM + p] = y + D * x;
    }
}

// ---------------------------------------------------------------------------
// y' = rms_norm(y * silu(z)) * nw, emitted as bf16 for the out_proj GEMM.
// One block per row, 256 threads x 6 elements over 1536 cols.
// ---------------------------------------------------------------------------
__global__ __launch_bounds__(256)
void gate_rms_kernel(const float* __restrict__ Y, const float* __restrict__ ZX,
                     const float* __restrict__ nw, bf16_t* __restrict__ OUT)
{
    __shared__ float red[256];
    const int row = blockIdx.x;
    const int tid = threadIdx.x;

    float vals[6];
    float ss = 0.0f;
    #pragma unroll
    for (int i = 0; i < 6; ++i) {
        const int c = tid + i * 256;
        const float y = Y[(size_t)row * DINNER + c];
        const float z = ZX[(size_t)row * NPROJP + c];
        const float v = y * z * sigmoidf_(z);
        vals[i] = v;
        ss += v * v;
    }
    red[tid] = ss;
    __syncthreads();
    for (int off = 128; off > 0; off >>= 1) {
        if (tid < off) red[tid] += red[tid + off];
        __syncthreads();
    }
    const float inv = rsqrtf(red[0] * (1.0f / DINNER) + 1e-5f);

    #pragma unroll
    for (int i = 0; i < 6; ++i) {
        const int c = tid + i * 256;
        OUT[(size_t)row * DINNER + c] = to_bf16(vals[i] * inv * nw[c]);
    }
}

// ---------------------------------------------------------------------------
// Output projection (768 -> 80). fp32, tiny.
// ---------------------------------------------------------------------------
__global__ void proj_out_kernel(const float* __restrict__ hidden, const float* __restrict__ W,
                                const float* __restrict__ b, float* __restrict__ pred)
{
    int idx = blockIdx.x * blockDim.x + threadIdx.x;
    if (idx >= NROWS * 80) return;
    int o = idx % 80;
    int row = idx / 80;
    float v = b[o];
    const float* hr = hidden + (size_t)row * DMODEL;
    #pragma unroll 8
    for (int k = 0; k < DMODEL; ++k) v += hr[k] * W[(size_t)k * 80 + o];
    pred[idx] = v;
}

// ---------------------------------------------------------------------------
// Host-side launch sequence
// ---------------------------------------------------------------------------
extern "C" void kernel_launch(void* const* d_in, const int* in_sizes, int n_in,
                              void* d_out, int out_size, void* d_ws, size_t ws_size,
                              hipStream_t stream)
{
    (void)in_sizes; (void)n_in; (void)out_size; (void)ws_size;

    const float* x          = (const float*)d_in[0];
    const int*   mask       = (const int*)  d_in[1];
    const float* proj_in_w  = (const float*)d_in[2];
    const float* proj_in_b  = (const float*)d_in[3];
    const float* mask_token = (const float*)d_in[4];
    const float* ln_w       = (const float*)d_in[5];
    const float* ln_b       = (const float*)d_in[6];
    const float* in_proj_w  = (const float*)d_in[7];
    const float* in_proj_b  = (const float*)d_in[8];
    const float* conv_w     = (const float*)d_in[9];
    const float* conv_b     = (const float*)d_in[10];
    const float* dt_bias    = (const float*)d_in[11];
    const float* A_log      = (const float*)d_in[12];
    const float* Dp         = (const float*)d_in[13];
    const float* ssm_norm_w = (const float*)d_in[14];
    const float* out_proj_w = (const float*)d_in[15];
    const float* fn_w       = (const float*)d_in[16];
    const float* fn_b       = (const float*)d_in[17];
    const float* proj_out_w = (const float*)d_in[18];
    const float* proj_out_b = (const float*)d_in[19];

    // --- workspace carving (256B-aligned) ---
    char* ws = (char*)d_ws;
    auto carve = [&](size_t bytes) -> char* {
        char* p = ws;
        ws += (bytes + 255) & ~(size_t)255;
        return p;
    };
    bf16_t* wt_in  = (bf16_t*)carve(4ull * NPROJP * DMODEL * sizeof(bf16_t)); // 4x (3200x768)
    bf16_t* wt_out = (bf16_t*)carve(4ull * DMODEL * DINNER * sizeof(bf16_t)); // 4x (768x1536)
    float*  H      = (float*) carve((size_t)NROWS * DMODEL * sizeof(float));
    bf16_t* HNB    = (bf16_t*)carve((size_t)NROWS * DMODEL * sizeof(bf16_t));
    float*  ZX     = (float*) carve((size_t)NROWS * NPROJP * sizeof(float));
    float*  XC     = (float*) carve((size_t)NROWS * CONVDIM * sizeof(float));
    float*  DTDA   = (float*) carve((size_t)NROWS * 2 * NHEADS * sizeof(float));
    float*  Y      = (float*) carve((size_t)NROWS * DINNER * sizeof(float));
    bf16_t* YBF    = (bf16_t*)carve((size_t)NROWS * DINNER * sizeof(bf16_t));

    float* out    = (float*)d_out;
    float* pred   = out;                       // (4,1024,80)
    float* hidden = out + (size_t)NROWS * 80;  // (4,1024,768)

    // --- per-launch bf16 weight conversion (pre-transposed, padded) ---
    for (int p = 0; p < 4; ++p) {
        const int n1 = NPROJP * DMODEL;
        convert_wt_kernel<<<(n1 + 255) / 256, 256, 0, stream>>>(
            in_proj_w + (size_t)p * DMODEL * NPROJ, wt_in + (size_t)p * NPROJP * DMODEL,
            DMODEL, NPROJ, NPROJP);
        const int n2 = DMODEL * DINNER;
        convert_wt_kernel<<<(n2 + 255) / 256, 256, 0, stream>>>(
            out_proj_w + (size_t)p * DINNER * DMODEL, wt_out + (size_t)p * DMODEL * DINNER,
            DINNER, DMODEL, DMODEL);
    }

    // --- stem: input projection + mask tokens ---
    proj_in_kernel<<<(NROWS * DMODEL + 255) / 256, 256, 0, stream>>>(
        x, mask, proj_in_w, proj_in_b, mask_token, H);

    // --- 2 layers x 2 directions ---
    for (int l = 0; l < 2; ++l) {
        layernorm768_kernel<<<NROWS, 256, 0, stream>>>(
            H, ln_w + l * DMODEL, ln_b + l * DMODEL, HNB, nullptr);
        for (int d = 0; d < 2; ++d) {
            const int p = l * 2 + d;
            // in_proj: (4096 x 3200) = HNB (4096x768) * Wt (3200x768)^T + bias
            dim3 g1(NROWS / 128, NPROJP / 128);
            gemm_bf16_kernel<<<g1, 256, 0, stream>>>(
                HNB, wt_in + (size_t)p * NPROJP * DMODEL, in_proj_b + p * NPROJ,
                ZX, NROWS, NPROJP, NPROJ, DMODEL, NPROJP, 0);

            dtda_kernel<<<(NROWS * NHEADS + 255) / 256, 256, 0, stream>>>(
                ZX, dt_bias + p * NHEADS, A_log + p * NHEADS, DTDA);

            conv_silu_kernel<<<(NROWS * CONVDIM + 255) / 256, 256, 0, stream>>>(
                ZX, conv_w + (size_t)p * CONVDIM * 7, conv_b + p * CONVDIM, XC, d);

            scan_kernel<<<dim3(4, 3), 512, 0, stream>>>(XC, DTDA, Dp + p * NHEADS, Y, d);

            gate_rms_kernel<<<NROWS, 256, 0, stream>>>(Y, ZX, ssm_norm_w + p * DINNER, YBF);

            // out_proj: H (residual) += YBF (4096x1536) * Wt (768x1536)^T
            dim3 g2(NROWS / 128, DMODEL / 128);
            gemm_bf16_kernel<<<g2, 256, 0, stream>>>(
                YBF, wt_out + (size_t)p * DMODEL * DINNER, nullptr,
                H, NROWS, DMODEL, DMODEL, DINNER, DMODEL, 1);
        }
    }

    // --- head: final LN (hidden, fp32 into d_out) + output projection ---
    layernorm768_kernel<<<NROWS, 256, 0, stream>>>(H, fn_w, fn_b, nullptr, hidden);
    proj_out_kernel<<<(NROWS * 80 + 255) / 256, 256, 0, stream>>>(
        hidden, proj_out_w, proj_out_b, pred);
}
